// ContropyLoss_82446192214711
// MI455X (gfx1250) — matrix operand order repaired and checked by reading, not compiled, not run on hardware
//
#include <hip/hip_runtime.h>
#include <math.h>

// Problem constants (match reference): N=4096, C=32000, D=1024
#define NROWS 4096
#define NCLS  32000
#define DFEAT 1024
#define ALPHA 0.1f
#define EPS_C 1e-8f

typedef __attribute__((ext_vector_type(16))) __bf16 v16bf;
typedef __attribute__((ext_vector_type(8)))  __bf16 v8bf;
typedef __attribute__((ext_vector_type(8)))  float  v8f;

// ---------------------------------------------------------------------------
// Kernel 1: per-row cross entropy via single-pass online softmax.
// One 256-thread block per row; logits read exactly once (memory-bound floor).
// Branchless online update: exactly one v_exp_f32 per element.
// ---------------------------------------------------------------------------
__global__ void ce_rows_kernel(const float* __restrict__ logits,
                               const int* __restrict__ labels,
                               float* __restrict__ ce_out) {
  const int row = blockIdx.x;
  const float* lr = logits + (size_t)row * NCLS;
  float m = -INFINITY, s = 0.f;
  for (int c = threadIdx.x * 4; c < NCLS; c += 256 * 4) {
    const float4 x4 = *(const float4*)(lr + c);
    const float xs[4] = {x4.x, x4.y, x4.z, x4.w};
#pragma unroll
    for (int e = 0; e < 4; ++e) {
      const float x  = xs[e];
      const float d  = x - m;
      const float ex = __expf(-fabsf(d));     // exp(-|x-m|): the only exp needed
      s = (d > 0.f) ? (s * ex + 1.f) : (s + ex);
      m = fmaxf(m, x);
    }
  }
  __shared__ float sm[256], ss[256];
  sm[threadIdx.x] = m; ss[threadIdx.x] = s;
  __syncthreads();
  for (int off = 128; off > 0; off >>= 1) {
    if (threadIdx.x < off) {
      const float m1 = sm[threadIdx.x],       s1 = ss[threadIdx.x];
      const float m2 = sm[threadIdx.x + off], s2 = ss[threadIdx.x + off];
      const float M = fmaxf(m1, m2);
      sm[threadIdx.x] = M;
      ss[threadIdx.x] = s1 * __expf(m1 - M) + s2 * __expf(m2 - M);
    }
    __syncthreads();
  }
  if (threadIdx.x == 0) {
    const float logZ = sm[0] + __logf(ss[0]);
    ce_out[row] = logZ - lr[labels[row]];
  }
}

// ---------------------------------------------------------------------------
// Kernel 2: per-row L2 norms of features (full f32 precision).
// ---------------------------------------------------------------------------
__global__ void norms_kernel(const float* __restrict__ f,
                             float* __restrict__ norms) {
  const int row = blockIdx.x;
  const float* fr = f + (size_t)row * DFEAT;
  const float4 x = *(const float4*)(fr + threadIdx.x * 4);  // 256*4 == 1024
  float s = x.x * x.x + x.y * x.y + x.z * x.z + x.w * x.w;
  __shared__ float red[256];
  red[threadIdx.x] = s;
  __syncthreads();
  for (int off = 128; off > 0; off >>= 1) {
    if (threadIdx.x < off) red[threadIdx.x] += red[threadIdx.x + off];
    __syncthreads();
  }
  if (threadIdx.x == 0) norms[row] = sqrtf(red[0]);
}

// ---------------------------------------------------------------------------
// Kernel 3: Gram tile GEMM (bf16 WMMA, f32 accumulate) fused with the
// cosine-hinge epilogue. 128x128 C tile per block; 8 waves x (32x64) strips.
// Ping-pong LDS double buffering: one barrier per K-step, next tile's global
// loads issued before the 8 WMMAs so HBM/L2 latency hides behind matrix work.
// Upper-triangular blocks only; off-diagonal blocks weighted 2x.
// ---------------------------------------------------------------------------
__device__ inline v8bf cvt8(const float4 a, const float4 b) {
  v8bf r;
  r[0] = (__bf16)a.x; r[1] = (__bf16)a.y; r[2] = (__bf16)a.z; r[3] = (__bf16)a.w;
  r[4] = (__bf16)b.x; r[5] = (__bf16)b.y; r[6] = (__bf16)b.z; r[7] = (__bf16)b.w;
  return r;
}

__global__ void __launch_bounds__(256)
gram_hinge_kernel(const float* __restrict__ f,
                  const float* __restrict__ norms,
                  float* __restrict__ partials) {
  const int bx = blockIdx.x, by = blockIdx.y;
  const int blk = by * 32 + bx;
  if (bx < by) {                                  // symmetric: skip lower triangle,
    if (threadIdx.x == 0) partials[blk] = 0.f;    // ws is poisoned -> must write 0
    return;
  }
  const int rowBase = by * 128, colBase = bx * 128;
  const int tid = threadIdx.x, lane = tid & 31, wid = tid >> 5;
  const int wm = wid & 3, wn = wid >> 2;          // 4x2 wave grid over 128x128

  __shared__ __align__(16) __bf16 As[2][128][40]; // stride 40 shorts keeps 16B align
  __shared__ __align__(16) __bf16 Bs[2][128][40];
  __shared__ float nI[128], nJ[128];
  __shared__ float red[256];

  if (tid < 128) nI[tid] = norms[rowBase + tid];
  else           nJ[tid - 128] = norms[colBase + (tid - 128)];

  v8f acc[2][4];
#pragma unroll
  for (int s2 = 0; s2 < 2; ++s2)
#pragma unroll
    for (int t2 = 0; t2 < 4; ++t2)
      acc[s2][t2] = (v8f){0.f, 0.f, 0.f, 0.f, 0.f, 0.f, 0.f, 0.f};

  const int lm = lane & 15;
  const int kh = (lane >> 4) << 3;          // A-frag K base within 32-K tile {0,8}
  const int kb = (lane >> 4) << 3;          // B-frag K base {0,8}
  const int r = tid >> 1, half = tid & 1;   // tile loaders: row r, 16-wide half

  const float* gaBase = f + (size_t)(rowBase + r) * DFEAT + half * 16;
  const float* gbBase = f + (size_t)(colBase + r) * DFEAT + half * 16;

  // ---- prologue: stage K-step 0 into buffer 0 ----
  float4 ra0 = ((const float4*)gaBase)[0], ra1 = ((const float4*)gaBase)[1];
  float4 ra2 = ((const float4*)gaBase)[2], ra3 = ((const float4*)gaBase)[3];
  float4 rb0 = ((const float4*)gbBase)[0], rb1 = ((const float4*)gbBase)[1];
  float4 rb2 = ((const float4*)gbBase)[2], rb3 = ((const float4*)gbBase)[3];
  *(v8bf*)&As[0][r][half * 16 + 0] = cvt8(ra0, ra1);
  *(v8bf*)&As[0][r][half * 16 + 8] = cvt8(ra2, ra3);
  *(v8bf*)&Bs[0][r][half * 16 + 0] = cvt8(rb0, rb1);
  *(v8bf*)&Bs[0][r][half * 16 + 8] = cvt8(rb2, rb3);

  const int NKT = DFEAT / 32;               // 32 K-steps
  for (int kt = 0; kt < NKT; ++kt) {
    const int cur = kt & 1;
    __syncthreads();                        // buf[cur] stores visible; buf[cur^1] free

    // ---- issue global loads for K-step kt+1 (latency hides behind WMMAs) ----
    const bool more = (kt + 1) < NKT;
    if (more) {
      const float* ga = gaBase + (kt + 1) * 32;
      const float* gb = gbBase + (kt + 1) * 32;
      ra0 = ((const float4*)ga)[0]; ra1 = ((const float4*)ga)[1];
      ra2 = ((const float4*)ga)[2]; ra3 = ((const float4*)ga)[3];
      rb0 = ((const float4*)gb)[0]; rb1 = ((const float4*)gb)[1];
      rb2 = ((const float4*)gb)[2]; rb3 = ((const float4*)gb)[3];
    }

    // ---- assemble fragments per ISA 16-bit A(16x32)/B(32x16) layouts ----
    v16bf af[2], bfr[4];
#pragma unroll
    for (int s2 = 0; s2 < 2; ++s2) {
      const int m = wm * 32 + s2 * 16 + lm;
      const v8bf lo = *(const v8bf*)&As[cur][m][kh];
      const v8bf hi = *(const v8bf*)&As[cur][m][kh + 16];
      af[s2] = __builtin_shufflevector(lo, hi, 0, 1, 2, 3, 4, 5, 6, 7,
                                       8, 9, 10, 11, 12, 13, 14, 15);
    }
#pragma unroll
    for (int t2 = 0; t2 < 4; ++t2) {
      const int n = wn * 64 + t2 * 16 + lm;
      const v8bf lo = *(const v8bf*)&Bs[cur][n][kb];
      const v8bf hi = *(const v8bf*)&Bs[cur][n][kb + 16];
      bfr[t2] = __builtin_shufflevector(lo, hi, 0, 1, 2, 3, 4, 5, 6, 7,
                                        8, 9, 10, 11, 12, 13, 14, 15);
    }
#pragma unroll
    for (int s2 = 0; s2 < 2; ++s2)
#pragma unroll
      for (int t2 = 0; t2 < 4; ++t2)
        acc[s2][t2] = __builtin_amdgcn_wmma_f32_16x16x32_bf16(
            false, af[s2], false, bfr[t2], (short)0, acc[s2][t2], false, false);

    // ---- stage K-step kt+1 into the other buffer ----
    if (more) {
      *(v8bf*)&As[cur ^ 1][r][half * 16 + 0] = cvt8(ra0, ra1);
      *(v8bf*)&As[cur ^ 1][r][half * 16 + 8] = cvt8(ra2, ra3);
      *(v8bf*)&Bs[cur ^ 1][r][half * 16 + 0] = cvt8(rb0, rb1);
      *(v8bf*)&Bs[cur ^ 1][r][half * 16 + 8] = cvt8(rb2, rb3);
    }
  }

  // ---- fused branchless epilogue: hinge(cos) with i==j masked out ----
  float part = 0.f;
#pragma unroll
  for (int s2 = 0; s2 < 2; ++s2) {
    float ni[8];
    const int mlBase = wm * 32 + s2 * 16 + ((lane >> 4) << 3);
#pragma unroll
    for (int v = 0; v < 8; ++v) ni[v] = nI[mlBase + v];    // hoist LDS reads
#pragma unroll
    for (int t2 = 0; t2 < 4; ++t2) {
      const int nl = wn * 64 + t2 * 16 + lm;
      const int j = colBase + nl;
      const float nj = nJ[nl];
#pragma unroll
      for (int v = 0; v < 8; ++v) {
        // C layout: M = v + 8*(lane>=16), N = lane&15
        const int i = rowBase + mlBase + v;
        const float den = fmaxf(ni[v] * nj, EPS_C);        // always > 0
        const float h = fmaxf(acc[s2][t2][v] * __builtin_amdgcn_rcpf(den), 0.f);
        part += (i != j) ? h : 0.f;                        // select, no branch
      }
    }
  }
  part *= (bx == by) ? 1.f : 2.f;     // symmetric doubling of off-diagonal blocks

  red[tid] = part;
  __syncthreads();
  for (int off = 128; off > 0; off >>= 1) {
    if (tid < off) red[tid] += red[tid + off];
    __syncthreads();
  }
  if (tid == 0) partials[blk] = red[0];
}

// ---------------------------------------------------------------------------
// Kernel 4: deterministic final reduction -> scalar loss.
// ---------------------------------------------------------------------------
__global__ void finalize_kernel(const float* __restrict__ ce,
                                const float* __restrict__ gp,
                                float* __restrict__ out) {
  __shared__ float red[256];
  __shared__ float ce_sum_s;
  float s = 0.f;
  for (int i = threadIdx.x; i < NROWS; i += 256) s += ce[i];
  red[threadIdx.x] = s;
  __syncthreads();
  for (int off = 128; off > 0; off >>= 1) {
    if (threadIdx.x < off) red[threadIdx.x] += red[threadIdx.x + off];
    __syncthreads();
  }
  if (threadIdx.x == 0) ce_sum_s = red[0];
  __syncthreads();

  float g = 0.f;
  for (int i = threadIdx.x; i < 1024; i += 256) g += gp[i];
  red[threadIdx.x] = g;
  __syncthreads();
  for (int off = 128; off > 0; off >>= 1) {
    if (threadIdx.x < off) red[threadIdx.x] += red[threadIdx.x + off];
    __syncthreads();
  }
  if (threadIdx.x == 0)
    out[0] = ce_sum_s / (float)NROWS +
             ALPHA * red[0] / ((float)NROWS * (float)(NROWS - 1));
}

// ---------------------------------------------------------------------------
extern "C" void kernel_launch(void* const* d_in, const int* in_sizes, int n_in,
                              void* d_out, int out_size, void* d_ws, size_t ws_size,
                              hipStream_t stream) {
  const float* logits   = (const float*)d_in[0];
  const int*   labels   = (const int*)d_in[1];
  const float* features = (const float*)d_in[2];
  float* out = (float*)d_out;

  float* ws    = (float*)d_ws;
  float* ce    = ws;           // [0, 4096)   per-row CE
  float* norms = ws + 4096;    // [4096,8192) per-row L2 norms
  float* gp    = ws + 8192;    // [8192,9216) per-block hinge partials (32x32)

  ce_rows_kernel<<<NROWS, 256, 0, stream>>>(logits, labels, ce);
  norms_kernel<<<NROWS, 256, 0, stream>>>(features, norms);
  dim3 grid(32, 32);
  gram_hinge_kernel<<<grid, 256, 0, stream>>>(features, norms, gp);
  finalize_kernel<<<1, 256, 0, stream>>>(ce, gp, out);
}
